// DSAMBlock_43301860278993
// MI455X (gfx1250) — matrix operand order repaired and checked by reading, not compiled, not run on hardware
//
#include <hip/hip_runtime.h>
#include <math.h>

typedef __attribute__((ext_vector_type(16))) _Float16 v16h;
typedef __attribute__((ext_vector_type(8)))  float    v8f;
typedef __attribute__((ext_vector_type(4)))  unsigned int u32x4;
typedef __attribute__((ext_vector_type(8)))  int      i32x8;
typedef __attribute__((ext_vector_type(4)))  int      i32x4;

#define N_IMG  4
#define C_ALL  64
#define C_HALF 32
#define HDIM   256
#define WDIM   256
#define PLANE  (HDIM * WDIM)                     // 65536
#define TENS   ((size_t)N_IMG * C_ALL * PLANE)   // 16777216 floats

#if defined(__HIP_DEVICE_COMPILE__) && __has_builtin(__builtin_amdgcn_tensor_load_to_lds)
#define USE_TDM 1
#else
#define USE_TDM 0
#endif

__device__ __forceinline__ int refl(int i, int L) {
    if (i < 0) i = -i;
    if (i >= L) i = 2 * L - 2 - i;
    return i;
}

// ---- Tensor Data Mover: DMA a 256x64-float tile (row stride 256 floats) into LDS ----
// D# per CDNA5 ISA ch.8: group0 = {count/lds_addr/global_addr/type}, group1 = dims/strides.
__device__ __forceinline__ void tdm_tile_256x64(const float* gsrc, unsigned ldsByteOff) {
#if USE_TDM
    unsigned long long ga = (unsigned long long)(size_t)gsrc;
    u32x4 g0;
    g0[0] = 1u;                                                  // count=1, user mode
    g0[1] = ldsByteOff;                                          // lds_addr [63:32]
    g0[2] = (unsigned)ga;                                        // global_addr[31:0]
    g0[3] = (unsigned)((ga >> 32) & 0x01FFFFFFu) | (2u << 30);   // addr[56:32], type=2
    i32x8 g1;
    g1[0] = 2 << 16;                                     // data_size=4B, no multicast
    g1[1] = (int)(256u << 16);                           // tensor_dim0(=256) lo16 @ [63:48]
    g1[2] = (int)(256u << 16);                           // dim0 hi16 | tensor_dim1(=256) lo16
    g1[3] = (int)(64u << 16);                            // dim1 hi16 | tile_dim0=64
    g1[4] = 256;                                         // tile_dim1=256, tile_dim2=0
    g1[5] = 256;                                         // tensor_dim0_stride lo32 = 256
    g1[6] = 0;                                           // stride hi | dim1_stride lo
    g1[7] = 0;
    i32x4 z4 = {0, 0, 0, 0};
#if __has_include(<hip/amd_detail/amd_gfx1250_TDM.h>)
    i32x8 z8 = {0, 0, 0, 0, 0, 0, 0, 0};
    __builtin_amdgcn_tensor_load_to_lds(g0, g1, z4, z4, z8, 0);  // therock 6-arg form
#else
    __builtin_amdgcn_tensor_load_to_lds(g0, g1, z4, z4, 0);      // ROCm 7.2 5-arg form
#endif
#else
    (void)gsrc; (void)ldsByteOff;
#endif
}

// ---------------- Pool-strip horizontal pass ----------------
template <int K>
__global__ void k_hori(const float* __restrict__ x, float* __restrict__ hout,
                       const float* __restrict__ prm) {
    __shared__ float srow[WDIM];
    __shared__ float red[256];
    int b = blockIdx.x;            // n*C*H + c*H + h
    int plane = b >> 8;
    int c = plane & 63;
    int w = threadIdx.x;
    size_t base = (size_t)b * WDIM;
    float xv = x[base + w];
    srow[w] = xv;
    __syncthreads();
    float l;
    if (K == 0) {
        red[w] = xv;
        __syncthreads();
        for (int s = 128; s > 0; s >>= 1) {
            if (w < s) red[w] += red[w + s];
            __syncthreads();
        }
        l = red[0] * (1.0f / WDIM);
    } else {
        const int p = K / 2;
        float acc = 0.f;
#pragma unroll
        for (int t = -p; t <= p; ++t) acc += srow[refl(w + t, WDIM)];
        l = acc * (1.0f / K);
    }
    float hl = prm[0 * C_ALL + c], hh = prm[1 * C_ALL + c];
    hout[base + w] = hl * l + (hh + 1.0f) * (xv - l);
}

// ---------------- Pool-strip vertical pass + accumulate into s ----------------
// 64 threads/block; 256x64 LDS tile staged via TDM (fallback: coalesced loads).
template <int K>
__global__ void k_vert(const float* __restrict__ hin, const float* __restrict__ x,
                       float* __restrict__ s, const float* __restrict__ prm,
                       const float* __restrict__ pg, const float* __restrict__ p7,
                       const float* __restrict__ p11, int addX) {
    __shared__ float tile[HDIM * 64];
    int b = blockIdx.x;
    int plane = b >> 2;            // n*64 + c
    int wt = b & 3;
    int c = plane & 63;
    int t = threadIdx.x;           // 0..63
    size_t pbase = (size_t)plane * PLANE + wt * 64;
#if USE_TDM
    if ((threadIdx.x >> 5) == 0) {
        tdm_tile_256x64(hin + pbase, (unsigned)(size_t)(void*)tile);
        __builtin_amdgcn_s_wait_tensorcnt(0);
    }
    __syncthreads();
#else
    for (int h = 0; h < HDIM; ++h)
        tile[h * 64 + t] = hin[pbase + (size_t)h * WDIM + t];
    __syncthreads();
#endif
    float vl = prm[2 * C_ALL + c], vh = prm[3 * C_ALL + c], g = prm[4 * C_ALL + c];
    float btot = 0.f;
    if (addX) btot = pg[5 * C_ALL + c] + p7[5 * C_ALL + c] + p11[5 * C_ALL + c];
    if (K == 0) {
        float m = 0.f;
        for (int h = 0; h < HDIM; ++h) m += tile[h * 64 + t];
        m *= (1.0f / HDIM);
        for (int h = 0; h < HDIM; ++h) {
            float hv = tile[h * 64 + t];
            float vo = vl * m + (vh + 1.0f) * (hv - m);
            size_t idx = pbase + (size_t)h * WDIM + t;
            float base = addX ? (x[idx] * btot) : s[idx];
            s[idx] = base + g * vo;
        }
    } else {
        const int p = K / 2;
        for (int h = 0; h < HDIM; ++h) {
            float acc = 0.f;
#pragma unroll
            for (int dt = -p; dt <= p; ++dt)
                acc += tile[refl(h + dt, HDIM) * 64 + t];
            float l = acc * (1.0f / K);
            float hv = tile[h * 64 + t];
            float vo = vl * l + (vh + 1.0f) * (hv - l);
            size_t idx = pbase + (size_t)h * WDIM + t;
            float base = addX ? (x[idx] * btot) : s[idx];
            s[idx] = base + g * vo;
        }
    }
}

// ---------------- 1x1 conv as WMMA GEMM ----------------
__global__ void k_conv_wmma(const float* __restrict__ s, const float* __restrict__ w,
                            const float* __restrict__ bias, float* __restrict__ out) {
    __shared__ _Float16 wl[C_ALL * C_ALL];   // spec_w as f16, [o][c]
    int tid = threadIdx.x;
    for (int i = tid; i < C_ALL * C_ALL; i += 256) wl[i] = (_Float16)w[i];
    __syncthreads();
    int wave = tid >> 5, lane = tid & 31;
    int tileId = blockIdx.x * 8 + wave;
    int col = lane & 15;
    int hi = lane >> 4;
    int gpix = tileId * 16 + col;
    int n = gpix >> 16;
    int p = gpix & 65535;
    v16h b0, b1;
    int kbase = hi * 16;
#pragma unroll
    for (int e = 0; e < 16; ++e) {
        b0[e] = (_Float16)s[((size_t)(n * C_ALL + kbase + e)) * PLANE + p];
        b1[e] = (_Float16)s[((size_t)(n * C_ALL + 32 + kbase + e)) * PLANE + p];
    }
    int row = lane & 15;
    int koff = hi * 16;
#pragma unroll
    for (int ot = 0; ot < 4; ++ot) {
        v16h a0, a1;
#pragma unroll
        for (int e = 0; e < 16; ++e) {
            int k = koff + e;
            a0[e] = wl[(ot * 16 + row) * C_ALL + k];
            a1[e] = wl[(ot * 16 + row) * C_ALL + 32 + k];
        }
        v8f c = {};
        c = __builtin_amdgcn_wmma_f32_16x16x32_f16(false, a0, false, b0, (short)0, c, false, false);
        c = __builtin_amdgcn_wmma_f32_16x16x32_f16(false, a1, false, b1, (short)0, c, false, false);
#pragma unroll
        for (int r = 0; r < 8; ++r) {
            int o = ot * 16 + hi * 8 + r;
            out[((size_t)(n * C_ALL + o)) * PLANE + p] = c[r] + bias[o];
        }
    }
}

// ---------------- global average pool ----------------
__global__ void k_gap(const float* __restrict__ in, float* __restrict__ gout) {
    __shared__ float red[256];
    int pl = blockIdx.x;
    int tid = threadIdx.x;
    const float* base = in + (size_t)pl * PLANE;
    float acc = 0.f;
    for (int i = tid; i < PLANE; i += 256) acc += base[i];
    red[tid] = acc;
    __syncthreads();
    for (int s = 128; s > 0; s >>= 1) { if (tid < s) red[tid] += red[tid + s]; __syncthreads(); }
    if (tid == 0) gout[pl] = red[0] * (1.0f / PLANE);
}

// ---------------- sigmoid(GAP @ W) tap filters ----------------
__global__ void k_filt(const float* __restrict__ gap, int gapStride, int cbase,
                       const float* __restrict__ w, int K, float* __restrict__ fout) {
    int idx = threadIdx.x;
    if (idx >= N_IMG * K) return;
    int n = idx / K, t = idx % K;
    float acc = 0.f;
    for (int c = 0; c < C_HALF; ++c)
        acc += gap[n * gapStride + cbase + c] * w[t * C_HALF + c];
    fout[idx] = 1.0f / (1.0f + expf(-acc));
}

// ---------------- horizontal strip attention ----------------
template <int K>
__global__ void k_stripH(const float* __restrict__ conv, int chanOff,
                         const float* __restrict__ filt, float* __restrict__ tout) {
    __shared__ float srow[WDIM];
    int b = blockIdx.x;
    int n = b >> 13;
    int cc = (b >> 8) & 31;
    int h = b & 255;
    int w = threadIdx.x;
    size_t inIdx = ((size_t)(n * C_ALL + chanOff + cc)) * PLANE + (size_t)h * WDIM + w;
    srow[w] = conv[inIdx];
    __syncthreads();
    const int p = K / 2;
    float acc = 0.f;
#pragma unroll
    for (int t = 0; t < K; ++t)
        acc += srow[refl(w + t - p, WDIM)] * filt[n * K + t];
    tout[((size_t)(n * C_HALF + cc)) * PLANE + (size_t)h * WDIM + w] = acc;
}

// ---------------- vertical strip attention + combine -> d_out ----------------
template <int K>
__global__ void k_stripV(const float* __restrict__ tin, const float* __restrict__ conv,
                         int chanOff, const float* __restrict__ filt,
                         const float* __restrict__ gb, float* __restrict__ out) {
    __shared__ float tile[HDIM * 64];
    int b = blockIdx.x;
    int plane = b >> 2;                  // n*32 + cc
    int wt = b & 3;
    int n = plane >> 5, cc = plane & 31;
    int t = threadIdx.x;
    size_t pbase = (size_t)plane * PLANE + wt * 64;
#if USE_TDM
    if ((threadIdx.x >> 5) == 0) {
        tdm_tile_256x64(tin + pbase, (unsigned)(size_t)(void*)tile);
        __builtin_amdgcn_s_wait_tensorcnt(0);
    }
    __syncthreads();
#else
    for (int h = 0; h < HDIM; ++h)
        tile[h * 64 + t] = tin[pbase + (size_t)h * WDIM + t];
    __syncthreads();
#endif
    float g0 = gb[cc], g1 = gb[C_HALF + cc];
    float f[K];
#pragma unroll
    for (int j = 0; j < K; ++j) f[j] = filt[n * K + j];
    const int p = K / 2;
    size_t xbase = ((size_t)(n * C_ALL + chanOff + cc)) * PLANE + wt * 64;
    for (int h = 0; h < HDIM; ++h) {
        float acc = 0.f;
#pragma unroll
        for (int dt = 0; dt < K; ++dt)
            acc += tile[refl(h + dt - p, HDIM) * 64 + t] * f[dt];
        float xv = conv[xbase + (size_t)h * WDIM + t];
        out[xbase + (size_t)h * WDIM + t] = g0 * acc + g1 * xv;
    }
}

extern "C" void kernel_launch(void* const* d_in, const int* in_sizes, int n_in,
                              void* d_out, int out_size, void* d_ws, size_t ws_size,
                              hipStream_t stream) {
    const float* x      = (const float*)d_in[0];
    const float* gp     = (const float*)d_in[1];
    const float* lp7    = (const float*)d_in[2];
    const float* lp11   = (const float*)d_in[3];
    const float* spec_w = (const float*)d_in[4];
    const float* spec_b = (const float*)d_in[5];
    const float* w11h   = (const float*)d_in[6];
    const float* w11w   = (const float*)d_in[7];
    const float* gb11   = (const float*)d_in[8];
    const float* w7h    = (const float*)d_in[9];
    const float* w7w    = (const float*)d_in[10];
    const float* gb7    = (const float*)d_in[11];
    float* out = (float*)d_out;
    float* ws  = (float*)d_ws;

    float* bufA = ws;                    // hori scratch, then conv output
    float* bufB = ws + TENS;             // s accumulator, then t1/t2
    float* sm   = ws + 2 * TENS;         // small scalars
    float* gap0   = sm;
    float* filtH1 = sm + 256;
    float* filtH2 = sm + 320;
    float* gap1b  = sm + 384;
    float* gap2b  = sm + 512;
    float* filtV1 = sm + 640;
    float* filtV2 = sm + 704;

    const int ROWS = N_IMG * C_ALL * HDIM;   // 65536

    k_hori<0><<<ROWS, 256, 0, stream>>>(x, bufA, gp);
    k_vert<0><<<1024, 64, 0, stream>>>(bufA, x, bufB, gp, gp, lp7, lp11, 1);
    k_hori<7><<<ROWS, 256, 0, stream>>>(x, bufA, lp7);
    k_vert<7><<<1024, 64, 0, stream>>>(bufA, x, bufB, lp7, gp, lp7, lp11, 0);
    k_hori<11><<<ROWS, 256, 0, stream>>>(x, bufA, lp11);
    k_vert<11><<<1024, 64, 0, stream>>>(bufA, x, bufB, lp11, gp, lp7, lp11, 0);

    k_conv_wmma<<<2048, 256, 0, stream>>>(bufB, spec_w, spec_b, bufA);

    float* t1 = bufB;
    float* t2 = bufB + (size_t)N_IMG * C_HALF * PLANE;
    k_gap<<<N_IMG * C_ALL, 256, 0, stream>>>(bufA, gap0);
    k_filt<<<1, 64, 0, stream>>>(gap0, C_ALL, 0,  w11h, 11, filtH1);
    k_filt<<<1, 64, 0, stream>>>(gap0, C_ALL, 32, w7h,  7,  filtH2);
    k_stripH<11><<<N_IMG * C_HALF * HDIM, 256, 0, stream>>>(bufA, 0,  filtH1, t1);
    k_stripH<7><<<N_IMG * C_HALF * HDIM, 256, 0, stream>>>(bufA, 32, filtH2, t2);
    k_gap<<<N_IMG * C_HALF, 256, 0, stream>>>(t1, gap1b);
    k_gap<<<N_IMG * C_HALF, 256, 0, stream>>>(t2, gap2b);
    k_filt<<<1, 64, 0, stream>>>(gap1b, C_HALF, 0, w11w, 11, filtV1);
    k_filt<<<1, 64, 0, stream>>>(gap2b, C_HALF, 0, w7w,  7,  filtV2);
    k_stripV<11><<<512, 64, 0, stream>>>(t1, bufA, 0,  filtV1, gb11, out);
    k_stripV<7><<<512, 64, 0, stream>>>(t2, bufA, 32, filtV2, gb7, out);

    (void)in_sizes; (void)n_in; (void)out_size; (void)ws_size;
}
